// AxialMSAEncoderBlock_87771951661328
// MI455X (gfx1250) — matrix operand order repaired and checked by reading, not compiled, not run on hardware
//
#include <hip/hip_runtime.h>
#include <hip/hip_bf16.h>
#include <math.h>

// ---------------------------------------------------------------------------
// AxialMSAEncoderBlock for MI455X (gfx1250, wave32, WMMA f32_16x16x32_f16)
// R=64, C=256, B=1, E=768, H=12, D=64, F=3072  -> T = 16384 tokens
// ---------------------------------------------------------------------------

typedef __attribute__((ext_vector_type(16))) _Float16 v16h;
typedef __attribute__((ext_vector_type(8)))  _Float16 v8h;
typedef __attribute__((ext_vector_type(8)))  float    v8f;
typedef __attribute__((ext_vector_type(4)))  float    v4f;
typedef __attribute__((ext_vector_type(4)))  int      v4i;

#define T_TOK 16384
#define E_DIM 768
#define F_DIM 3072
#define NH    12
#define RROWS 64
#define CCOLS 256

#if defined(__has_builtin)
#if __has_builtin(__builtin_amdgcn_global_load_async_to_lds_b128)
#define HAVE_ASYNC_B128 1
#endif
#endif

__device__ __forceinline__ v16h cat16(v8h a, v8h b) {
  return __builtin_shufflevector(a, b, 0,1,2,3,4,5,6,7,8,9,10,11,12,13,14,15);
}
__device__ __forceinline__ v8h ld8h(const _Float16* p) { return *(const v8h*)p; }
__device__ __forceinline__ v8f wmma32(v16h a, v16h b, v8f c) {
  return __builtin_amdgcn_wmma_f32_16x16x32_f16(false, a, false, b, (short)0, c,
                                                false, false);
}

// 16-byte global -> LDS copy; async (ASYNCcnt / bypasses VGPRs) when available
__device__ __forceinline__ void copy16_g2l(void* lds, const void* g) {
#if HAVE_ASYNC_B128
  __builtin_amdgcn_global_load_async_to_lds_b128(
      (__attribute__((address_space(1))) v4i*)g,
      (__attribute__((address_space(3))) v4i*)lds, 0, 0);
#else
  *(v8h*)lds = *(const v8h*)g;
#endif
}
__device__ __forceinline__ void wait_async0() {
#if HAVE_ASYNC_B128
#if __has_builtin(__builtin_amdgcn_s_wait_asynccnt)
  __builtin_amdgcn_s_wait_asynccnt(0);
#else
  asm volatile("s_wait_asynccnt 0x0" ::: "memory");
#endif
#endif
}

// branch-free e^x via v_exp_f32 (inputs bounded in our uses)
__device__ __forceinline__ float fast_exp(float x) {
#if __has_builtin(__builtin_amdgcn_exp2f)
  return __builtin_amdgcn_exp2f(x * 1.44269504088896340736f);
#else
  return __expf(x);
#endif
}
// tanh-approx GELU (matches jax.nn.gelu approximate=True), branch-free
__device__ __forceinline__ float gelu_f(float x) {
  float y = 0.7978845608028654f * (x + 0.044715f * x * x * x);
  float t = fast_exp(-2.0f * fabsf(y));          // e^{-2|y|}
  float th = (1.0f - t) / (1.0f + t);
  th = __builtin_copysignf(th, y);
  return 0.5f * x * (1.0f + th);
}

// ---------------------------------------------------------------------------
// LayerNorm over E=768, one block (256 threads) per token, f16 output
// ---------------------------------------------------------------------------
__global__ __launch_bounds__(256) void ln_f16_kernel(
    const float* __restrict__ x, const float* __restrict__ s,
    const float* __restrict__ b, _Float16* __restrict__ out) {
  const int t = blockIdx.x;
  const int tid = threadIdx.x;
  const float* xp = x + (size_t)t * E_DIM;
  float v0 = xp[tid], v1 = xp[tid + 256], v2 = xp[tid + 512];
  float sum = v0 + v1 + v2;
  float sq  = v0 * v0 + v1 * v1 + v2 * v2;
#pragma unroll
  for (int m = 16; m >= 1; m >>= 1) {
    sum += __shfl_xor(sum, m, 32);
    sq  += __shfl_xor(sq,  m, 32);
  }
  __shared__ float red[16];
  const int wid = tid >> 5, lane = tid & 31;
  if (lane == 0) { red[wid] = sum; red[wid + 8] = sq; }
  __syncthreads();
  float ts = 0.f, tq = 0.f;
#pragma unroll
  for (int i = 0; i < 8; ++i) { ts += red[i]; tq += red[i + 8]; }
  const float mu   = ts * (1.0f / 768.0f);
  const float var  = tq * (1.0f / 768.0f) - mu * mu;
  const float rstd = rsqrtf(var + 1e-6f);
  _Float16* op = out + (size_t)t * E_DIM;
  op[tid]       = (_Float16)((v0 - mu) * rstd * s[tid]       + b[tid]);
  op[tid + 256] = (_Float16)((v1 - mu) * rstd * s[tid + 256] + b[tid + 256]);
  op[tid + 512] = (_Float16)((v2 - mu) * rstd * s[tid + 512] + b[tid + 512]);
}

// ---------------------------------------------------------------------------
// Generic WMMA GEMM: out[M,N] = act((A_f16[M,K] @ W_f32[K,N] + bias) * alpha)
// Block tile 128x128, Ktile 32, 8 waves (2x4), wave tile 64x32 (4x2 frags).
// Double-buffered LDS; A tile staged with GLOBAL_LOAD_ASYNC_TO_LDS_B128 so
// the next tile's copy overlaps the current tile's WMMAs.
// ---------------------------------------------------------------------------
template <bool GELU, bool OUTH, bool OUTF>
__global__ __launch_bounds__(256) void gemm_wmma_kernel(
    const _Float16* __restrict__ A, int lda,
    const float* __restrict__ W, int ldw,
    const float* __restrict__ bias, float alpha, int K,
    _Float16* __restrict__ outH, float* __restrict__ outF, int ldo) {
  __shared__ __align__(16) _Float16 As[2][128 * 40];  // [m][k], pad 8 halves
  __shared__ __align__(16) _Float16 Bs[2][128 * 40];  // [n][k] transposed
  const int tid = threadIdx.x;
  const int lane = tid & 31, w = tid >> 5;
  const int l16 = lane & 15, lhi = lane >> 4;
  const int waveM = w & 1, waveN = w >> 1;
  const int m0 = blockIdx.y * 128;
  const int n0 = blockIdx.x * 128;
  const int kiters = K >> 5;

  auto loadA = [&](int kt, int buf) {   // 128x32 f16, async -> LDS
    const int k0 = kt * 32;
#pragma unroll
    for (int it = 0; it < 2; ++it) {
      int m  = (tid >> 2) + it * 64;
      int ch = (tid & 3) * 8;
      copy16_g2l(&As[buf][m * 40 + ch],
                 A + (size_t)(m0 + m) * lda + k0 + ch);
    }
  };
  auto loadB = [&](int kt, int buf) {   // 32x128 f32 -> f16 transposed
    const int k0 = kt * 32;
#pragma unroll
    for (int it = 0; it < 4; ++it) {
      int idx = it * 1024 + tid * 4;
      int kk = idx >> 7;
      int nn = idx & 127;
      v4f wv = *(const v4f*)(W + (size_t)(k0 + kk) * ldw + n0 + nn);
#pragma unroll
      for (int j = 0; j < 4; ++j) Bs[buf][(nn + j) * 40 + kk] = (_Float16)wv[j];
    }
  };

  v8f acc[4][2] = {};
  loadA(0, 0);
  loadB(0, 0);
  wait_async0();
  __syncthreads();

  for (int kt = 0; kt < kiters; ++kt) {
    const int buf = kt & 1;
    if (kt + 1 < kiters) {              // overlap next tile with WMMAs
      loadA(kt + 1, buf ^ 1);
      loadB(kt + 1, buf ^ 1);
      __builtin_prefetch(W + (size_t)(kt * 32 + 64 + (tid >> 3)) * ldw + n0 +
                             (tid & 7) * 16, 0, 1);
    }
    v16h af[4], bf[2];
#pragma unroll
    for (int mt = 0; mt < 4; ++mt) {
      int m = waveM * 64 + mt * 16 + l16;
      int kb = lhi * 16;
      af[mt] = cat16(*(const v8h*)(&As[buf][m * 40 + kb]),
                     *(const v8h*)(&As[buf][m * 40 + kb + 8]));
    }
#pragma unroll
    for (int nt = 0; nt < 2; ++nt) {
      int n = waveN * 32 + nt * 16 + l16;
      int kb = lhi * 16;
      bf[nt] = cat16(*(const v8h*)(&Bs[buf][n * 40 + kb]),
                     *(const v8h*)(&Bs[buf][n * 40 + kb + 8]));
    }
#pragma unroll
    for (int mt = 0; mt < 4; ++mt)
#pragma unroll
      for (int nt = 0; nt < 2; ++nt)
        acc[mt][nt] = wmma32(af[mt], bf[nt], acc[mt][nt]);
    wait_async0();
    __syncthreads();
  }

#pragma unroll
  for (int mt = 0; mt < 4; ++mt) {
#pragma unroll
    for (int nt = 0; nt < 2; ++nt) {
      int gn = n0 + waveN * 32 + nt * 16 + l16;
      float bv = bias[gn];
#pragma unroll
      for (int r2 = 0; r2 < 8; ++r2) {
        int gm = m0 + waveM * 64 + mt * 16 + r2 + 8 * lhi;
        float val = (acc[mt][nt][r2] + bv) * alpha;
        if constexpr (GELU) val = gelu_f(val);
        size_t off = (size_t)gm * ldo + gn;
        if constexpr (OUTH) outH[off] = (_Float16)val;
        if constexpr (OUTF) outF[off] = val;
      }
    }
  }
}

// ---------------------------------------------------------------------------
// Tied-row attention scores + softmax:
// S[h,i,j] = sum_{r,d} q[r,i,h,d] * k[r,j,h,d]; P = softmax_j(S)  (256x256/head)
// grid = 12 heads * 2; 8 waves, each wave owns one 16-row i-tile.
// K slices staged cooperatively into LDS (async, double buffered): all 8
// waves share the same B fragments, so this cuts L2 traffic 8x and removes
// the per-WMMA full load-wait.
// ---------------------------------------------------------------------------
__global__ __launch_bounds__(256) void row_attn_scores_kernel(
    const _Float16* __restrict__ q, const _Float16* __restrict__ k,
    _Float16* __restrict__ P) {
  const int head = blockIdx.x >> 1;
  const int half = blockIdx.x & 1;
  const int tid = threadIdx.x;
  const int lane = tid & 31, w = tid >> 5;
  const int l16 = lane & 15, lhi = lane >> 4;
  const int i0 = (half * 8 + w) * 16;
  const int h64 = head * 64;

  // 64 j-rows x 64 d halves per chunk (stride 72), double buffered = 18 KB
  __shared__ __align__(16) _Float16 Kl[2][64 * 72];

  // chunk step = r*4 + qc stages k[r*256 + qc*64 + j][h64 + d], j=0..63
  auto stageK = [&](int step, int buf) {
    int r = step >> 2, qc = step & 3;
    int j = tid >> 2;                  // 0..63
    int dch = (tid & 3) * 16;          // 0,16,32,48 halves
    const _Float16* kp =
        k + ((size_t)(r * CCOLS + qc * 64 + j) * E_DIM + h64 + dch);
    copy16_g2l(&Kl[buf][j * 72 + dch], kp);
    copy16_g2l(&Kl[buf][j * 72 + dch + 8], kp + 8);
  };

  v8f acc[16] = {};
  stageK(0, 0);
  wait_async0();
  __syncthreads();

  for (int r = 0; r < RROWS; ++r) {
    v16h af[2];
#pragma unroll
    for (int dc = 0; dc < 2; ++dc) {
      const _Float16* qp =
          q + ((size_t)(r * CCOLS + i0 + l16) * E_DIM + h64 + dc * 32 + lhi * 16);
      af[dc] = cat16(ld8h(qp), ld8h(qp + 8));
    }
#pragma unroll
    for (int qc = 0; qc < 4; ++qc) {
      const int step = r * 4 + qc;
      const int buf = step & 1;
      if (step + 1 < RROWS * 4) stageK(step + 1, buf ^ 1);
#pragma unroll
      for (int dc = 0; dc < 2; ++dc) {
        int d0 = dc * 32;
#pragma unroll
        for (int jt4 = 0; jt4 < 4; ++jt4) {
          const _Float16* bp = &Kl[buf][(jt4 * 16 + l16) * 72 + d0 + lhi * 16];
          v16h bf = cat16(*(const v8h*)bp, *(const v8h*)(bp + 8));
          acc[qc * 4 + jt4] = wmma32(af[dc], bf, acc[qc * 4 + jt4]);
        }
      }
      wait_async0();
      __syncthreads();
    }
  }
  // softmax over j (256 cols): reduce across 16 frags and 16 lanes (half-wave)
#pragma unroll
  for (int r2 = 0; r2 < 8; ++r2) {
    float m = -3.0e38f;
#pragma unroll
    for (int jt = 0; jt < 16; ++jt) m = fmaxf(m, acc[jt][r2]);
#pragma unroll
    for (int s = 8; s >= 1; s >>= 1) m = fmaxf(m, __shfl_xor(m, s, 32));
    float p[16], sum = 0.f;
#pragma unroll
    for (int jt = 0; jt < 16; ++jt) { p[jt] = fast_exp(acc[jt][r2] - m); sum += p[jt]; }
#pragma unroll
    for (int s = 8; s >= 1; s >>= 1) sum += __shfl_xor(sum, s, 32);
    float inv = 1.0f / sum;
    int row = i0 + r2 + 8 * lhi;
#pragma unroll
    for (int jt = 0; jt < 16; ++jt)
      P[(size_t)head * 65536 + (size_t)row * 256 + jt * 16 + l16] =
          (_Float16)(p[jt] * inv);
  }
}

// ---------------------------------------------------------------------------
// Tied-row attention PV: ctx[r,i,h,d] = sum_j P[h,i,j] * v[r,j,h,d]
// grid = 12 heads * 64 rows; V tile transposed via LDS per 32-j chunk.
// ---------------------------------------------------------------------------
__global__ __launch_bounds__(256) void row_attn_pv_kernel(
    const _Float16* __restrict__ P, const _Float16* __restrict__ v,
    _Float16* __restrict__ ctx) {
  const int head = blockIdx.x >> 6;
  const int r = blockIdx.x & 63;
  __shared__ __align__(16) _Float16 Vt[64 * 40];   // [d][j_local], pad 8
  const int tid = threadIdx.x;
  const int lane = tid & 31, w = tid >> 5;
  const int l16 = lane & 15, lhi = lane >> 4;
  const int h64 = head * 64;
  const int i0 = w * 32;

  v8f acc[2][4] = {};
  for (int kj = 0; kj < 8; ++kj) {
    const int j0 = kj * 32;
    __syncthreads();
    {
      int jl = tid >> 3;
      int dch = (tid & 7) * 8;
      v8h vv = ld8h(v + ((size_t)(r * CCOLS + j0 + jl) * E_DIM + h64 + dch));
#pragma unroll
      for (int e = 0; e < 8; ++e) Vt[(dch + e) * 40 + jl] = vv[e];
    }
    __syncthreads();

    v16h af[2], bf[4];
#pragma unroll
    for (int mt = 0; mt < 2; ++mt) {
      const _Float16* pp = P + (size_t)head * 65536 +
                           (size_t)(i0 + mt * 16 + l16) * 256 + j0 + lhi * 16;
      af[mt] = cat16(ld8h(pp), ld8h(pp + 8));
    }
#pragma unroll
    for (int nt = 0; nt < 4; ++nt) {
      int n = nt * 16 + l16;
      int kb = lhi * 16;
      bf[nt] = cat16(*(const v8h*)(&Vt[n * 40 + kb]),
                     *(const v8h*)(&Vt[n * 40 + kb + 8]));
    }
#pragma unroll
    for (int mt = 0; mt < 2; ++mt)
#pragma unroll
      for (int nt = 0; nt < 4; ++nt)
        acc[mt][nt] = wmma32(af[mt], bf[nt], acc[mt][nt]);
  }
#pragma unroll
  for (int mt = 0; mt < 2; ++mt)
#pragma unroll
    for (int nt = 0; nt < 4; ++nt)
#pragma unroll
      for (int r2 = 0; r2 < 8; ++r2) {
        int i = i0 + mt * 16 + r2 + 8 * lhi;
        int d = nt * 16 + l16;
        ctx[(size_t)(r * CCOLS + i) * E_DIM + h64 + d] =
            (_Float16)acc[mt][nt][r2];
      }
}

// ---------------------------------------------------------------------------
// Column attention, fully fused per (head, column): 64x64 scores over rows,
// softmax, PV. 128 threads = 4 waves, one 16-row i-tile per wave.
// ---------------------------------------------------------------------------
__global__ __launch_bounds__(128) void col_attn_kernel(
    const _Float16* __restrict__ q, const _Float16* __restrict__ k,
    const _Float16* __restrict__ v, _Float16* __restrict__ ctx) {
  const int head = blockIdx.x >> 8;
  const int c = blockIdx.x & 255;
  const int tid = threadIdx.x;
  const int lane = tid & 31, w = tid >> 5;
  const int l16 = lane & 15, lhi = lane >> 4;
  const int h64 = head * 64;
  __shared__ __align__(16) _Float16 Vt[64 * 72];       // [d][j], pad 8
  __shared__ __align__(16) _Float16 Pl[4][16 * 72];    // per-wave P, pad 8

  {  // stage V transposed: 64 rows (j) x 64 (d)
    int j = tid >> 1;
    int dbase = (tid & 1) * 32;
    const _Float16* vp = v + ((size_t)(j * CCOLS + c) * E_DIM + h64 + dbase);
#pragma unroll
    for (int cc = 0; cc < 4; ++cc) {
      v8h vv = ld8h(vp + cc * 8);
#pragma unroll
      for (int e = 0; e < 8; ++e) Vt[(dbase + cc * 8 + e) * 72 + j] = vv[e];
    }
  }
  __syncthreads();

  const int i0 = w * 16;
  v8f s[4] = {};
#pragma unroll
  for (int dc = 0; dc < 2; ++dc) {
    int d0 = dc * 32;
    const _Float16* qp =
        q + ((size_t)((i0 + l16) * CCOLS + c) * E_DIM + h64 + d0 + lhi * 16);
    v16h af = cat16(ld8h(qp), ld8h(qp + 8));
#pragma unroll
    for (int jt = 0; jt < 4; ++jt) {
      const _Float16* kp =
          k + ((size_t)((jt * 16 + l16) * CCOLS + c) * E_DIM + h64 + d0 + lhi * 16);
      v16h bf = cat16(ld8h(kp), ld8h(kp + 8));
      s[jt] = wmma32(af, bf, s[jt]);
    }
  }
  // softmax over 64 cols, write P into wave-local LDS (A-fragment friendly)
#pragma unroll
  for (int r2 = 0; r2 < 8; ++r2) {
    float m = -3.0e38f;
#pragma unroll
    for (int jt = 0; jt < 4; ++jt) m = fmaxf(m, s[jt][r2]);
#pragma unroll
    for (int sh = 8; sh >= 1; sh >>= 1) m = fmaxf(m, __shfl_xor(m, sh, 32));
    float p[4], sum = 0.f;
#pragma unroll
    for (int jt = 0; jt < 4; ++jt) { p[jt] = fast_exp(s[jt][r2] - m); sum += p[jt]; }
#pragma unroll
    for (int sh = 8; sh >= 1; sh >>= 1) sum += __shfl_xor(sum, sh, 32);
    float inv = 1.0f / sum;
    int row = r2 + 8 * lhi;
#pragma unroll
    for (int jt = 0; jt < 4; ++jt)
      Pl[w][row * 72 + jt * 16 + l16] = (_Float16)(p[jt] * inv);
  }
  __syncthreads();

  v8f o[4] = {};
#pragma unroll
  for (int kc = 0; kc < 2; ++kc) {
    int j0 = kc * 32;
    const _Float16* pp = &Pl[w][l16 * 72 + j0 + lhi * 16];
    v16h af = cat16(*(const v8h*)pp, *(const v8h*)(pp + 8));
#pragma unroll
    for (int nt = 0; nt < 4; ++nt) {
      int n = nt * 16 + l16;
      const _Float16* bp = &Vt[n * 72 + j0 + lhi * 16];
      v16h bf = cat16(*(const v8h*)bp, *(const v8h*)(bp + 8));
      o[nt] = wmma32(af, bf, o[nt]);
    }
  }
#pragma unroll
  for (int nt = 0; nt < 4; ++nt)
#pragma unroll
    for (int r2 = 0; r2 < 8; ++r2) {
      int i = i0 + r2 + 8 * lhi;
      int d = nt * 16 + l16;
      ctx[(size_t)(i * CCOLS + c) * E_DIM + h64 + d] = (_Float16)o[nt][r2];
    }
}

// ---------------------------------------------------------------------------
// Host-side launch sequence
// ---------------------------------------------------------------------------
extern "C" void kernel_launch(void* const* d_in, const int* in_sizes, int n_in,
                              void* d_out, int out_size, void* d_ws, size_t ws_size,
                              hipStream_t stream) {
  const float* x     = (const float*)d_in[0];
  const float* ln1_s = (const float*)d_in[1];
  const float* ln1_b = (const float*)d_in[2];
  const float* r_wq  = (const float*)d_in[3];
  const float* r_bq  = (const float*)d_in[4];
  const float* r_wk  = (const float*)d_in[5];
  const float* r_bk  = (const float*)d_in[6];
  const float* r_wv  = (const float*)d_in[7];
  const float* r_bv  = (const float*)d_in[8];
  const float* r_wo  = (const float*)d_in[9];
  const float* r_bo  = (const float*)d_in[10];
  const float* ln2_s = (const float*)d_in[11];
  const float* ln2_b = (const float*)d_in[12];
  const float* c_wq  = (const float*)d_in[13];
  const float* c_bq  = (const float*)d_in[14];
  const float* c_wk  = (const float*)d_in[15];
  const float* c_bk  = (const float*)d_in[16];
  const float* c_wv  = (const float*)d_in[17];
  const float* c_bv  = (const float*)d_in[18];
  const float* c_wo  = (const float*)d_in[19];
  const float* c_bo  = (const float*)d_in[20];
  const float* ln3_s = (const float*)d_in[21];
  const float* ln3_b = (const float*)d_in[22];
  const float* f_w1  = (const float*)d_in[23];
  const float* f_b1  = (const float*)d_in[24];
  const float* f_w2  = (const float*)d_in[25];
  const float* f_b2  = (const float*)d_in[26];

  char* ws = (char*)d_ws;
  const size_t SZ_TE16 = (size_t)T_TOK * E_DIM * 2;     // 25,165,824 B
  _Float16* h16   = (_Float16*)(ws);
  _Float16* q16   = (_Float16*)(ws + SZ_TE16);
  _Float16* k16   = (_Float16*)(ws + 2 * SZ_TE16);
  _Float16* v16p  = (_Float16*)(ws + 3 * SZ_TE16);
  _Float16* ctx16 = (_Float16*)(ws + 4 * SZ_TE16);
  _Float16* ff1   = q16;                                 // reuse q..ctx (T*F f16)
  float*    Xbuf  = (float*)(ws + 5 * SZ_TE16);          // T*E f32
  _Float16* Prow  = (_Float16*)(ws + 5 * SZ_TE16 + (size_t)T_TOK * E_DIM * 4);

  const dim3 gGemmE(E_DIM / 128, T_TOK / 128);   // (6,128)
  const dim3 gGemmF(F_DIM / 128, T_TOK / 128);   // (24,128)
  const float rowScale = 1.0f / 64.0f;           // D^-0.5 / sqrt(R)
  const float colScale = 0.125f;                 // D^-0.5

  // ---- tied-row self-attention ----
  ln_f16_kernel<<<T_TOK, 256, 0, stream>>>(x, ln1_s, ln1_b, h16);
  gemm_wmma_kernel<false, true, false><<<gGemmE, 256, 0, stream>>>(
      h16, E_DIM, r_wq, E_DIM, r_bq, rowScale, E_DIM, q16, nullptr, E_DIM);
  gemm_wmma_kernel<false, true, false><<<gGemmE, 256, 0, stream>>>(
      h16, E_DIM, r_wk, E_DIM, r_bk, 1.0f, E_DIM, k16, nullptr, E_DIM);
  gemm_wmma_kernel<false, true, false><<<gGemmE, 256, 0, stream>>>(
      h16, E_DIM, r_wv, E_DIM, r_bv, 1.0f, E_DIM, v16p, nullptr, E_DIM);
  row_attn_scores_kernel<<<NH * 2, 256, 0, stream>>>(q16, k16, Prow);
  row_attn_pv_kernel<<<NH * RROWS, 256, 0, stream>>>(Prow, v16p, ctx16);
  gemm_wmma_kernel<false, false, true><<<gGemmE, 256, 0, stream>>>(
      ctx16, E_DIM, r_wo, E_DIM, r_bo, 1.0f, E_DIM, nullptr, Xbuf, E_DIM);

  // ---- column self-attention ----
  ln_f16_kernel<<<T_TOK, 256, 0, stream>>>(Xbuf, ln2_s, ln2_b, h16);
  gemm_wmma_kernel<false, true, false><<<gGemmE, 256, 0, stream>>>(
      h16, E_DIM, c_wq, E_DIM, c_bq, colScale, E_DIM, q16, nullptr, E_DIM);
  gemm_wmma_kernel<false, true, false><<<gGemmE, 256, 0, stream>>>(
      h16, E_DIM, c_wk, E_DIM, c_bk, 1.0f, E_DIM, k16, nullptr, E_DIM);
  gemm_wmma_kernel<false, true, false><<<gGemmE, 256, 0, stream>>>(
      h16, E_DIM, c_wv, E_DIM, c_bv, 1.0f, E_DIM, v16p, nullptr, E_DIM);
  col_attn_kernel<<<NH * CCOLS, 128, 0, stream>>>(q16, k16, v16p, ctx16);
  gemm_wmma_kernel<false, false, true><<<gGemmE, 256, 0, stream>>>(
      ctx16, E_DIM, c_wo, E_DIM, c_bo, 1.0f, E_DIM, nullptr, Xbuf, E_DIM);

  // ---- FFN with GELU ----
  ln_f16_kernel<<<T_TOK, 256, 0, stream>>>(Xbuf, ln3_s, ln3_b, h16);
  gemm_wmma_kernel<true, true, false><<<gGemmF, 256, 0, stream>>>(
      h16, E_DIM, f_w1, F_DIM, f_b1, 1.0f, E_DIM, ff1, nullptr, F_DIM);
  gemm_wmma_kernel<false, false, true><<<gGemmE, 256, 0, stream>>>(
      ff1, F_DIM, f_w2, E_DIM, f_b2, 1.0f, F_DIM, nullptr, (float*)d_out, E_DIM);
}
